// Attention_2439541424688
// MI455X (gfx1250) — matrix operand order repaired and checked by reading, not compiled, not run on hardware
//
#include <hip/hip_runtime.h>

// Problem constants (match reference)
#define NB 4      // batch
#define CH 256    // channels
#define NN 4096   // positions
#define DD 32     // CH/8  (theta/phi dim)
#define GG 128    // CH/2  (g dim)

typedef __attribute__((ext_vector_type(16))) __bf16   bf16x16;
typedef __attribute__((ext_vector_type(2)))  __bf16   bf16x2;
typedef __attribute__((ext_vector_type(8)))  float    f32x8;
typedef __attribute__((ext_vector_type(2)))  float    f32x2;
typedef __attribute__((ext_vector_type(8)))  unsigned u32x8;

// Hardware f32->bf16 pair conversion (lowers to v_cvt_pk_bf16_f32-class ops)
static __device__ __forceinline__ unsigned packbf(float a, float b) {
  f32x2 v = {a, b};
  return __builtin_bit_cast(unsigned, __builtin_convertvector(v, bf16x2));
}
static __device__ __forceinline__ unsigned short bf16b(float f) {
  return __builtin_bit_cast(unsigned short, (__bf16)f);
}
static __device__ __forceinline__ f32x8 wmma32(u32x8 a, u32x8 b, f32x8 c) {
  return __builtin_amdgcn_wmma_f32_16x16x32_bf16(
      false, __builtin_bit_cast(bf16x16, a),
      false, __builtin_bit_cast(bf16x16, b),
      (short)0, c, false, false);
}

// A-matrix (16x32 bf16) per-lane K base: lanes 0-15 hold K 0-15, lanes 16-31 hold K 16-31
#define KBA(v) (hf * 16 + 2 * (v))
// B-matrix (32x16 bf16) per-lane K base: lanes 0-15 hold K 0-15, lanes 16-31 hold K 16-31
#define KBB(v) (hf * 16 + 2 * (v))

// ---------------------------------------------------------------------------
// Kernel 0: convert Wo (f32 [256][128]) -> bf16
// ---------------------------------------------------------------------------
__global__ void cvt_wo_kernel(const float* __restrict__ Wo,
                              unsigned short* __restrict__ Wo16, int n) {
  int i = blockIdx.x * 256 + threadIdx.x;
  if (i < n) Wo16[i] = bf16b(Wo[i]);
}

// ---------------------------------------------------------------------------
// Kernel 1: projections theta/phi (position-major [b][n][32] bf16)
//           and g (channel-major [b][c][n] bf16), via WMMA over K=256.
// One wave per (b, 16-position tile).
// ---------------------------------------------------------------------------
__global__ __launch_bounds__(256) void proj_kernel(
    const float* __restrict__ x, const float* __restrict__ Wt,
    const float* __restrict__ Wp, const float* __restrict__ Wg,
    unsigned short* __restrict__ theta, unsigned short* __restrict__ phi,
    unsigned short* __restrict__ gT) {
  const int wave = blockIdx.x * 8 + (threadIdx.x >> 5);
  const int lane = threadIdx.x & 31;
  const int b = wave >> 8;
  const int n0 = (wave & 255) * 16;
  const int lr = lane & 15, hf = lane >> 4;

  const float* xb = x + (size_t)b * CH * NN;

  // B operand: x tile (K=channels, N=positions), 8 chunks of K=32
  u32x8 Bx[8];
  for (int kc = 0; kc < 8; ++kc) {
#pragma unroll
    for (int v = 0; v < 8; ++v) {
      int kb = kc * 32 + KBB(v);
      float a0 = xb[(size_t)kb * NN + n0 + lr];
      float a1 = xb[(size_t)(kb + 1) * NN + n0 + lr];
      Bx[kc][v] = packbf(a0, a1);
    }
  }

  for (int ot = 0; ot < 12; ++ot) {
    const float* W;
    int orow;
    if (ot < 2)       { W = Wt; orow = ot * 16; }
    else if (ot < 4)  { W = Wp; orow = (ot - 2) * 16; }
    else              { W = Wg; orow = (ot - 4) * 16; }

    f32x8 acc = {};
    for (int kc = 0; kc < 8; ++kc) {
      u32x8 Aw;
#pragma unroll
      for (int v = 0; v < 8; ++v) {
        int kb = kc * 32 + KBA(v);
        f32x2 w2 = *(const f32x2*)(W + (size_t)(orow + lr) * CH + kb);
        Aw[v] = packbf(w2.x, w2.y);
      }
      acc = wmma32(Aw, Bx[kc], acc);
    }
    // D layout: lane column n = lr, VGPR j -> output row orow + j + 8*hf
    if (ot < 4) {
      unsigned short* dst = (ot < 2) ? theta : phi;
#pragma unroll
      for (int j = 0; j < 8; ++j) {
        int d = orow + j + 8 * hf;
        dst[((size_t)b * NN + n0 + lr) * DD + d] = bf16b(acc[j]);
      }
    } else {
#pragma unroll
      for (int j = 0; j < 8; ++j) {
        int c = orow + j + 8 * hf;
        gT[((size_t)b * GG + c) * NN + n0 + lr] = bf16b(acc[j]);
      }
    }
  }
}

// ---------------------------------------------------------------------------
// Kernel 2: fused flash attention + output projection + residual.
// One wave per (b, 16-query tile); streams all 4096 keys in steps of 32.
// ---------------------------------------------------------------------------
__global__ __launch_bounds__(256) void attn_kernel(
    const unsigned short* __restrict__ theta16,
    const unsigned short* __restrict__ phi16,
    const unsigned short* __restrict__ gT16,
    const unsigned short* __restrict__ Wo16,
    const float* __restrict__ x, const float* __restrict__ gamma_p,
    float* __restrict__ out) {
  const int wave = blockIdx.x * 8 + (threadIdx.x >> 5);
  const int lane = threadIdx.x & 31;
  const int b = wave >> 8;
  const int n0 = (wave & 255) * 16;
  const int lr = lane & 15, hf = lane >> 4;

  const unsigned* thU = (const unsigned*)theta16;
  const unsigned* phU = (const unsigned*)phi16;
  const unsigned* gU  = (const unsigned*)gT16;
  const unsigned* woU = (const unsigned*)Wo16;

  // B operand: theta queries (K=32 channels, N=16 queries). Loaded once.
  u32x8 Bth;
#pragma unroll
  for (int v = 0; v < 8; ++v)
    Bth[v] = thU[(((size_t)b * NN + n0 + lr) * DD + KBB(v)) >> 1];

  f32x8 acc[8];
#pragma unroll
  for (int t = 0; t < 8; ++t) acc[t] = (f32x8){};
  float rmax = -INFINITY, rsum = 0.0f;

  for (int m0 = 0; m0 < NN; m0 += 32) {
    // Scores for 32 keys: A = phi key rows, B = theta.  D[m][q], column q = lr.
    u32x8 Aph;
#pragma unroll
    for (int v = 0; v < 8; ++v)
      Aph[v] = phU[(((size_t)b * NN + m0 + lr) * DD + KBA(v)) >> 1];
    f32x8 S0 = wmma32(Aph, Bth, (f32x8){});
#pragma unroll
    for (int v = 0; v < 8; ++v)
      Aph[v] = phU[(((size_t)b * NN + m0 + 16 + lr) * DD + KBA(v)) >> 1];
    f32x8 S1 = wmma32(Aph, Bth, (f32x8){});

    // Online softmax over the 32 keys.  Lane (lr,hf) holds keys
    // {8hf+j} (S0) and {16+8hf+j} (S1); its partner (xor 16) holds the rest.
    float tmax = S0[0];
#pragma unroll
    for (int j = 1; j < 8; ++j) tmax = fmaxf(tmax, S0[j]);
#pragma unroll
    for (int j = 0; j < 8; ++j) tmax = fmaxf(tmax, S1[j]);
    tmax = fmaxf(tmax, __shfl_xor(tmax, 16, 32));
    float nmax = fmaxf(rmax, tmax);
    float scale = __expf(rmax - nmax);
    rsum *= scale;
#pragma unroll
    for (int t = 0; t < 8; ++t)
#pragma unroll
      for (int j = 0; j < 8; ++j) acc[t][j] *= scale;

    float psum = 0.0f;
    f32x8 P0, P1;
#pragma unroll
    for (int j = 0; j < 8; ++j) { P0[j] = __expf(S0[j] - nmax); psum += P0[j]; }
#pragma unroll
    for (int j = 0; j < 8; ++j) { P1[j] = __expf(S1[j] - nmax); psum += P1[j]; }
    psum += __shfl_xor(psum, 16, 32);
    rsum += psum;
    rmax = nmax;

    // Repack P (C-layout) into B-layout for the attn matmul.
    // B needs P[K = 16hf + 2v .. +1][q]; one xor-16 exchange supplies the
    // half not owned by this lane.
    unsigned pk0[4], pk1[4];
#pragma unroll
    for (int v = 0; v < 4; ++v) {
      pk0[v] = packbf(P0[2 * v], P0[2 * v + 1]);
      pk1[v] = packbf(P1[2 * v], P1[2 * v + 1]);
    }
    u32x8 BP;
#pragma unroll
    for (int v = 0; v < 4; ++v) {
      unsigned x0 = __shfl_xor((int)pk0[v], 16, 32);
      unsigned x1 = __shfl_xor((int)pk1[v], 16, 32);
      BP[v]     = hf ? x1     : pk0[v];
      BP[v + 4] = hf ? pk1[v] : x0;
    }

    // attn[c][q] += g[c][m] * P[m][q] over 8 c-tiles of 16
#pragma unroll
    for (int ct = 0; ct < 8; ++ct) {
      u32x8 Ag;
#pragma unroll
      for (int v = 0; v < 8; ++v)
        Ag[v] = gU[(((size_t)b * GG + ct * 16 + lr) * NN + m0 + KBA(v)) >> 1];
      acc[ct] = wmma32(Ag, BP, acc[ct]);
    }
  }

  // Normalize by softmax denominator (per query column = per lane).
  float inv = 1.0f / rsum;
#pragma unroll
  for (int t = 0; t < 8; ++t)
#pragma unroll
    for (int j = 0; j < 8; ++j) acc[t][j] *= inv;

  // Repack attn accumulators into B operands for the Wo projection (K=128
  // in 4 chunks of 32 channels).  Same xor-16 exchange pattern as BP.
  u32x8 Bc[4];
#pragma unroll
  for (int k0 = 0; k0 < 4; ++k0) {
    unsigned a0[4], a1[4];
#pragma unroll
    for (int v = 0; v < 4; ++v) {
      a0[v] = packbf(acc[2 * k0][2 * v],     acc[2 * k0][2 * v + 1]);
      a1[v] = packbf(acc[2 * k0 + 1][2 * v], acc[2 * k0 + 1][2 * v + 1]);
    }
#pragma unroll
    for (int v = 0; v < 4; ++v) {
      unsigned x0 = __shfl_xor((int)a0[v], 16, 32);
      unsigned x1 = __shfl_xor((int)a1[v], 16, 32);
      Bc[k0][v]     = hf ? x1    : a0[v];
      Bc[k0][v + 4] = hf ? a1[v] : x0;
    }
  }

  const float gm = gamma_p[0];
  for (int ot = 0; ot < 16; ++ot) {
    f32x8 Do = {};
#pragma unroll
    for (int k0 = 0; k0 < 4; ++k0) {
      u32x8 Aw;
#pragma unroll
      for (int v = 0; v < 8; ++v)
        Aw[v] = woU[((size_t)(ot * 16 + lr) * GG + k0 * 32 + KBA(v)) >> 1];
      Do = wmma32(Aw, Bc[k0], Do);
    }
    // D: lane column q = lr, row och = 16*ot + j + 8*hf.  gamma*o + x.
#pragma unroll
    for (int j = 0; j < 8; ++j) {
      int och = ot * 16 + j + 8 * hf;
      size_t idx = ((size_t)b * CH + och) * NN + n0 + lr;
      out[idx] = gm * Do[j] + x[idx];
    }
  }
}

// ---------------------------------------------------------------------------
extern "C" void kernel_launch(void* const* d_in, const int* in_sizes, int n_in,
                              void* d_out, int out_size, void* d_ws,
                              size_t ws_size, hipStream_t stream) {
  const float* x  = (const float*)d_in[0];
  const float* Wt = (const float*)d_in[1];
  const float* Wp = (const float*)d_in[2];
  const float* Wg = (const float*)d_in[3];
  const float* Wo = (const float*)d_in[4];
  const float* gm = (const float*)d_in[5];
  float* out = (float*)d_out;

  // Workspace layout (bf16 shorts): theta | phi | gT | Wo16  (~6.4 MB)
  unsigned short* theta = (unsigned short*)d_ws;
  unsigned short* phi   = theta + (size_t)NB * NN * DD;
  unsigned short* gT    = phi   + (size_t)NB * NN * DD;
  unsigned short* Wo16  = gT    + (size_t)NB * GG * NN;

  cvt_wo_kernel<<<(CH * GG + 255) / 256, 256, 0, stream>>>(Wo, Wo16, CH * GG);
  // 1024 waves = 128 blocks x 8 waves, one wave per (b, 16-position tile)
  proj_kernel<<<128, 256, 0, stream>>>(x, Wt, Wp, Wg, theta, phi, gT);
  attn_kernel<<<128, 256, 0, stream>>>(theta, phi, gT, Wo16, x, gm, out);
}
